// EncodeBlock_71262097375682
// MI455X (gfx1250) — matrix operand order repaired and checked by reading, not compiled, not run on hardware
//
#include <hip/hip_runtime.h>
#include <cstdint>
#include <math.h>

#define D_EMB 1024
#define NHEAD 8
#define HD    128
#define BSZ   16
#define SEQ   512
#define TCH   64               // SEQ / N_AGENTS
#define BS    (BSZ*SEQ)        // 8192 tokens
#define BH    (BSZ*NHEAD)      // 128 (b,h) batches

typedef __bf16 bf16;
typedef __attribute__((ext_vector_type(16))) __bf16 bf16x16;
typedef __attribute__((ext_vector_type(8)))  float  f32x8;
typedef long long i64;

// ---------------- helpers ----------------

__device__ __forceinline__ bf16 f2bf(float f) {
  union { float f; unsigned u; } v; v.f = f;
  unsigned r = (v.u + 0x7FFFu + ((v.u >> 16) & 1u)) >> 16;   // round-to-nearest-even
  union { unsigned short s; bf16 b; } o; o.s = (unsigned short)r;
  return o.b;
}

__device__ __forceinline__ f32x8 wmma_bf16(bf16x16 a, bf16x16 b, f32x8 c) {
  return __builtin_amdgcn_wmma_f32_16x16x32_bf16(false, a, false, b,
                                                 (short)0, c, false, false);
}

// CDNA5 async global->LDS copy (16 bytes per lane), tracked by ASYNCcnt.
__device__ __forceinline__ void async_copy16(unsigned lds_off, const void* g) {
  asm volatile("global_load_async_to_lds_b128 %0, %1, off"
               :: "v"(lds_off), "v"((unsigned long long)(uintptr_t)g)
               : "memory");
}
__device__ __forceinline__ void wait_async0() {
#if __has_builtin(__builtin_amdgcn_s_wait_asynccnt)
  __builtin_amdgcn_s_wait_asynccnt(0);
#else
  asm volatile("s_wait_asynccnt 0x0" ::: "memory");
#endif
}

// One 16x16 f32 tile of A(MxK,row-major) @ Bt(NxK,row-major)^T, operands in global.
// A/Bt point at the tile's first row. K multiple of 32.
__device__ __forceinline__ f32x8 wmma_dot(const bf16* __restrict__ A, int lda,
                                          const bf16* __restrict__ Bt, int ldb,
                                          int K, int lane) {
  f32x8 acc = {};
  const int r  = lane & 15;
  const int ka = (lane >> 4) * 16;   // A frag: lanes 0-15 K 0-15, lanes 16-31 K 16-31
  const int kb = (lane >> 4) * 16;   // B frag: lanes 0-15 K 0-15, lanes 16-31 K 16-31
  const bf16* ap = A + (i64)r * lda + ka;
  const bf16* bp = Bt + (i64)r * ldb + kb;
  for (int k0 = 0; k0 < K; k0 += 32) {
    bf16x16 a, b;
#pragma unroll
    for (int i = 0; i < 8; ++i) { a[i] = ap[k0 + i]; a[i + 8] = ap[k0 + 8 + i]; }
#pragma unroll
    for (int i = 0; i < 16; ++i) { b[i] = bp[k0 + i]; }
    acc = wmma_bf16(a, b, acc);
  }
  return acc;
}

// ---------------- prep kernels ----------------

__global__ void k_transpose_bf16(const float* __restrict__ src, bf16* __restrict__ dst,
                                 int rows, int cols) {
  i64 boff = (i64)blockIdx.y * rows * cols;
  i64 idx = (i64)blockIdx.x * blockDim.x + threadIdx.x;
  if (idx >= (i64)rows * cols) return;
  int c = (int)(idx % cols), r = (int)(idx / cols);
  dst[boff + (i64)c * rows + r] = f2bf(src[boff + idx]);
}

__global__ void k_seg(const unsigned char* __restrict__ dones, int* __restrict__ seg) {
  int b = blockIdx.x * blockDim.x + threadIdx.x;
  if (b >= BSZ) return;
  int acc = 0;
  for (int s = 0; s < SEQ; ++s) { acc += dones[b * SEQ + s] ? 1 : 0; seg[b * SEQ + s] = acc; }
}

__global__ void k_pe(const float* __restrict__ x, const int* __restrict__ step,
                     bf16* __restrict__ xpe, bf16* __restrict__ xb) {
  i64 tok = blockIdx.x;
  int b = (int)(tok / SEQ), s = (int)(tok % SEQ);
  float pos = (float)step[b * SEQ + s];
  const float* xr = x + tok * D_EMB;
  bf16* xo = xpe + tok * D_EMB;
  bf16* xn = xb + tok * D_EMB;
  for (int d = threadIdx.x; d < D_EMB; d += blockDim.x) {
    int i = (d < D_EMB / 2) ? d : d - D_EMB / 2;
    float inv = exp2f(-(2.0f * (float)i / (float)D_EMB) * 13.287712379549449f); // log2(1e4)
    float ang = pos * inv;
    float pe = (d < D_EMB / 2) ? sinf(ang) : cosf(ang);
    float xv = xr[d];
    xo[d] = f2bf(xv + pe);
    xn[d] = f2bf(xv);
  }
}

// ---------------- LDS-staged register-blocked WMMA GEMM ----------------
// C[M,N] = A[M,K] @ Bt[N,K]^T. Block = 256 thr = 8 waves, block tile 128x64,
// wave tile 32x32 (2x2 WMMA frags). A/B panels staged via async global->LDS.
// Requires M%128==0, N%64==0, K%32==0.
#define BM 128
#define BN 64
#define BK 32

__global__ void __launch_bounds__(256)
k_gemm_lds(const bf16* __restrict__ A, int lda,
           const bf16* __restrict__ Bt, int ldb,
           float* __restrict__ C, int ldc,
           int N, int K) {
  __shared__ __align__(16) bf16 As[BM * BK];
  __shared__ __align__(16) bf16 Bs[BN * BK];
  const int tid = threadIdx.x;
  const int lane = tid & 31, wave = tid >> 5;
  const int wm = wave >> 1, wn = wave & 1;          // 4 x 2 wave grid
  const int tiles_n = N / BN;
  const int bm = blockIdx.x / tiles_n, bn = blockIdx.x % tiles_n;
  const int gm = bm * BM, gn = bn * BN;

  // staging: A = 512 16B chunks (2/thread), B = 256 16B chunks (1/thread)
  const int ar0 = (tid * 2) >> 2,     as0 = ((tid * 2) & 3) * 8;
  const int ar1 = (tid * 2 + 1) >> 2, as1 = ((tid * 2 + 1) & 3) * 8;
  const int br  = tid >> 2,           bsg = (tid & 3) * 8;
  const unsigned lA0 = (unsigned)(uintptr_t)&As[ar0 * BK + as0];
  const unsigned lA1 = (unsigned)(uintptr_t)&As[ar1 * BK + as1];
  const unsigned lB  = (unsigned)(uintptr_t)&Bs[br * BK + bsg];

  const int fr = lane & 15;
  const int ka = (lane >> 4) * 16;
  const int kb = (lane >> 4) * 16;

  f32x8 acc00 = {}, acc01 = {}, acc10 = {}, acc11 = {};

  for (int k0 = 0; k0 < K; k0 += BK) {
    async_copy16(lA0, A + (i64)(gm + ar0) * lda + k0 + as0);
    async_copy16(lA1, A + (i64)(gm + ar1) * lda + k0 + as1);
    async_copy16(lB,  Bt + (i64)(gn + br) * ldb + k0 + bsg);
    wait_async0();
    __syncthreads();

    const bf16* a0p = &As[(wm * 32 + fr) * BK + ka];
    const bf16* a1p = &As[(wm * 32 + 16 + fr) * BK + ka];
    const bf16* b0p = &Bs[(wn * 32 + fr) * BK + kb];
    const bf16* b1p = &Bs[(wn * 32 + 16 + fr) * BK + kb];
    bf16x16 a0, a1, b0, b1;
#pragma unroll
    for (int i = 0; i < 8; ++i) {
      a0[i] = a0p[i]; a0[i + 8] = a0p[8 + i];
      a1[i] = a1p[i]; a1[i + 8] = a1p[8 + i];
    }
#pragma unroll
    for (int i = 0; i < 16; ++i) { b0[i] = b0p[i]; b1[i] = b1p[i]; }
    acc00 = wmma_bf16(a0, b0, acc00);
    acc01 = wmma_bf16(a0, b1, acc01);
    acc10 = wmma_bf16(a1, b0, acc10);
    acc11 = wmma_bf16(a1, b1, acc11);
    __syncthreads();
  }

  const int n = lane & 15, mb = (lane >> 4) * 8;
  const int r0 = gm + wm * 32 + mb, r1 = r0 + 16;
  const int c0 = gn + wn * 32 + n,  c1 = c0 + 16;
#pragma unroll
  for (int rr = 0; rr < 8; ++rr) {
    C[(i64)(r0 + rr) * ldc + c0] = acc00[rr];
    C[(i64)(r0 + rr) * ldc + c1] = acc01[rr];
    C[(i64)(r1 + rr) * ldc + c0] = acc10[rr];
    C[(i64)(r1 + rr) * ldc + c1] = acc11[rr];
  }
}

// ---------------- q/k/v scatter into head-major bf16 operands ----------------
__global__ void k_qkv_scatter(const float* __restrict__ qf, const float* __restrict__ kf,
                              const float* __restrict__ vf, const int* __restrict__ seg,
                              bf16* __restrict__ qb, bf16* __restrict__ qdb,
                              bf16* __restrict__ kb, bf16* __restrict__ vT,
                              bf16* __restrict__ kTw) {
  i64 idx = (i64)blockIdx.x * blockDim.x + threadIdx.x;
  if (idx >= (i64)BS * D_EMB) return;
  int dfull = (int)(idx % D_EMB);
  i64 bs = idx / D_EMB;
  int s = (int)(bs % SEQ), b = (int)(bs / SEQ);
  int h = dfull / HD, d = dfull % HD;
  float kappa = 1.0f - exp2f(-5.0f - (float)h);
  float lk = log2f(kappa);
  int t = s >> 3;                                         // s / N_AGENTS
  i64 o_sh = (((i64)b * NHEAD + h) * SEQ + s) * HD + d;   // (b,h,s,d)
  i64 o_tr = (((i64)b * NHEAD + h) * HD + d) * SEQ + s;   // (b,h,d,s)
  float q = qf[idx], k = kf[idx], v = vf[idx];
  qb[o_sh] = f2bf(q);
  qdb[o_sh] = f2bf(q * exp2f(lk * (float)(t + 1)));       // q * kappa^(t+1)
  kb[o_sh] = f2bf(k);
  vT[o_tr] = f2bf(v);
  const int* segb = seg + b * SEQ;
  float lastok = (segb[s] == segb[SEQ - 1]) ? 1.0f : 0.0f;
  kTw[o_tr] = f2bf(k * exp2f(lk * (float)(TCH - t)) * lastok);  // k * kappa^(T-t) * last_ok
}

// ---------------- attention kernels (fused epilogues, direct-global WMMA) ----------------

__global__ void __launch_bounds__(256)
k_scores(const bf16* __restrict__ qb, const bf16* __restrict__ kb,
         const int* __restrict__ seg, bf16* __restrict__ sc) {
  int lane = threadIdx.x & 31, wave = threadIdx.x >> 5;
  int bh = blockIdx.y, b = bh / NHEAD, h = bh % NHEAD;
  int tile = blockIdx.x * 8 + wave;
  int tm = tile >> 5, tn = tile & 31;
  const bf16* A = qb + (i64)bh * SEQ * HD + (i64)tm * 16 * HD;
  const bf16* Bt = kb + (i64)bh * SEQ * HD + (i64)tn * 16 * HD;
  f32x8 acc = wmma_dot(A, HD, Bt, HD, HD, lane);
  float kappa = 1.0f - exp2f(-5.0f - (float)h);
  float lk = log2f(kappa);
  const float scale = 0.08838834764831845f;        // 1/sqrt(128)
  const int* segb = seg + b * SEQ;
  int n = lane & 15, mb = (lane >> 4) * 8;
  int j = tn * 16 + n, tj = j >> 3, sj = segb[j];
  bf16* outp = sc + (i64)bh * SEQ * SEQ;
#pragma unroll
  for (int rr = 0; rr < 8; ++rr) {
    int i = tm * 16 + mb + rr;
    float dm = exp2f(lk * fabsf((float)((i >> 3) - tj)));
    float val = (segb[i] == sj) ? acc[rr] * scale * dm : 0.0f;
    outp[(i64)i * SEQ + j] = f2bf(val);
  }
}

__global__ void __launch_bounds__(256)
k_inner(const bf16* __restrict__ sc, const bf16* __restrict__ vT,
        float* __restrict__ ret) {
  int lane = threadIdx.x & 31, wave = threadIdx.x >> 5;
  int bh = blockIdx.y, b = bh / NHEAD, h = bh % NHEAD;
  int tile = blockIdx.x * 8 + wave;
  int tm = tile >> 3, tn = tile & 7;
  const bf16* A = sc + (i64)bh * SEQ * SEQ + (i64)tm * 16 * SEQ;
  const bf16* Bt = vT + (i64)bh * HD * SEQ + (i64)tn * 16 * SEQ;
  f32x8 acc = wmma_dot(A, SEQ, Bt, SEQ, SEQ, lane);
  float* outp = ret + (i64)b * SEQ * D_EMB + h * HD;
  int n = lane & 15, mb = (lane >> 4) * 8;
  int e = tn * 16 + n;
#pragma unroll
  for (int rr = 0; rr < 8; ++rr)
    outp[(i64)(tm * 16 + mb + rr) * D_EMB + e] = acc[rr];
}

__global__ void __launch_bounds__(256)
k_cross(const bf16* __restrict__ qdb, const bf16* __restrict__ hT,
        const int* __restrict__ seg, float* __restrict__ ret) {
  int lane = threadIdx.x & 31, wave = threadIdx.x >> 5;
  int bh = blockIdx.y, b = bh / NHEAD, h = bh % NHEAD;
  int tile = blockIdx.x * 8 + wave;
  int tm = tile >> 3, tn = tile & 7;
  const bf16* A = qdb + (i64)bh * SEQ * HD + (i64)tm * 16 * HD;
  const bf16* Bt = hT + (i64)bh * HD * HD + (i64)tn * 16 * HD;
  f32x8 acc = wmma_dot(A, HD, Bt, HD, HD, lane);
  const int* segb = seg + b * SEQ;
  float* outp = ret + (i64)b * SEQ * D_EMB + h * HD;
  int n = lane & 15, mb = (lane >> 4) * 8;
  int e = tn * 16 + n;
#pragma unroll
  for (int rr = 0; rr < 8; ++rr) {
    int s = tm * 16 + mb + rr;
    if (segb[s] == 0) outp[(i64)s * D_EMB + e] += acc[rr];
  }
}

__global__ void __launch_bounds__(256)
k_nexth(const bf16* __restrict__ kTw, const bf16* __restrict__ vT,
        const float* __restrict__ hstate, const int* __restrict__ seg,
        float* __restrict__ nh) {
  int lane = threadIdx.x & 31, wave = threadIdx.x >> 5;
  int bh = blockIdx.y, b = bh / NHEAD, h = bh % NHEAD;
  int tile = blockIdx.x * 8 + wave;
  int tm = tile >> 3, tn = tile & 7;
  const bf16* A = kTw + (i64)bh * HD * SEQ + (i64)tm * 16 * SEQ;
  const bf16* Bt = vT + (i64)bh * HD * SEQ + (i64)tn * 16 * SEQ;
  f32x8 acc = wmma_dot(A, SEQ, Bt, SEQ, SEQ, lane);
  float kappa = 1.0f - exp2f(-5.0f - (float)h);
  float cd = exp2f((float)TCH * log2f(kappa));
  float nodone = (seg[b * SEQ + SEQ - 1] == 0) ? 1.0f : 0.0f;
  const float* hs = hstate + (i64)bh * HD * HD;
  float* outp = nh + (i64)bh * HD * HD;
  int n = lane & 15, mb = (lane >> 4) * 8;
  int e = tn * 16 + n;
#pragma unroll
  for (int rr = 0; rr < 8; ++rr) {
    int d = tm * 16 + mb + rr;
    outp[(i64)d * HD + e] = acc[rr] + cd * nodone * hs[(i64)d * HD + e];
  }
}

// ---------------- elementwise / norms ----------------

__global__ void k_conv_bf16(const float* __restrict__ src, bf16* __restrict__ dst, i64 n) {
  i64 idx = (i64)blockIdx.x * blockDim.x + threadIdx.x;
  if (idx < n) dst[idx] = f2bf(src[idx]);
}

__global__ void k_rms1(const float* __restrict__ x, const float* __restrict__ y,
                       const float* __restrict__ w, float* __restrict__ x1f,
                       bf16* __restrict__ x1b) {
  i64 row = blockIdx.x;
  const float* xr = x + row * D_EMB;
  const float* yr = y + row * D_EMB;
  __shared__ float red[8];
  float vals[4], ss = 0.f;
#pragma unroll
  for (int i = 0; i < 4; ++i) {
    int d = threadIdx.x + i * 256;
    float v = xr[d] + yr[d];
    vals[i] = v; ss += v * v;
  }
  for (int offn = 16; offn > 0; offn >>= 1) ss += __shfl_down(ss, offn, 32);
  if ((threadIdx.x & 31) == 0) red[threadIdx.x >> 5] = ss;
  __syncthreads();
  if (threadIdx.x == 0) { float t = 0.f; for (int i = 0; i < 8; ++i) t += red[i]; red[0] = t; }
  __syncthreads();
  float inv = rsqrtf(red[0] / (float)D_EMB + 1e-6f);
  float* o = x1f + row * D_EMB;
  bf16* ob = x1b + row * D_EMB;
#pragma unroll
  for (int i = 0; i < 4; ++i) {
    int d = threadIdx.x + i * 256;
    float r = vals[i] * inv * w[d];
    o[d] = r; ob[d] = f2bf(r);
  }
}

__global__ void k_final(const float* __restrict__ x1, const float* __restrict__ g,
                        const float* __restrict__ p, const float* __restrict__ w,
                        float* __restrict__ outp) {
  i64 row = blockIdx.x;
  const float* xr = x1 + row * D_EMB;
  const float* gr = g + row * D_EMB;
  const float* pr = p + row * D_EMB;
  __shared__ float red[8];
  float vals[4], ss = 0.f;
#pragma unroll
  for (int i = 0; i < 4; ++i) {
    int d = threadIdx.x + i * 256;
    float gv = gr[d];
    float v = xr[d] + (gv / (1.0f + expf(-gv))) * pr[d];
    vals[i] = v; ss += v * v;
  }
  for (int offn = 16; offn > 0; offn >>= 1) ss += __shfl_down(ss, offn, 32);
  if ((threadIdx.x & 31) == 0) red[threadIdx.x >> 5] = ss;
  __syncthreads();
  if (threadIdx.x == 0) { float t = 0.f; for (int i = 0; i < 8; ++i) t += red[i]; red[0] = t; }
  __syncthreads();
  float inv = rsqrtf(red[0] / (float)D_EMB + 1e-6f);
  float* o = outp + row * D_EMB;
#pragma unroll
  for (int i = 0; i < 4; ++i) {
    int d = threadIdx.x + i * 256;
    o[d] = vals[i] * inv * w[d];
  }
}

// ---------------- host orchestration ----------------

extern "C" void kernel_launch(void* const* d_in, const int* in_sizes, int n_in,
                              void* d_out, int out_size, void* d_ws, size_t ws_size,
                              hipStream_t stream) {
  (void)in_sizes; (void)n_in; (void)out_size; (void)ws_size;
  const float* x      = (const float*)d_in[0];
  const float* hstate = (const float*)d_in[1];
  const unsigned char* dones = (const unsigned char*)d_in[2];  // jnp bool -> 1 byte
  const int*   step   = (const int*)d_in[3];
  const float* wln1   = (const float*)d_in[4];
  const float* wln2   = (const float*)d_in[5];
  const float* Wq = (const float*)d_in[6];
  const float* Wk = (const float*)d_in[7];
  const float* Wv = (const float*)d_in[8];
  const float* Wo = (const float*)d_in[9];
  const float* Wg = (const float*)d_in[10];
  const float* Wp = (const float*)d_in[11];

  float* out   = (float*)d_out;                 // (B,S,D) f32
  float* nhout = out + (i64)BS * D_EMB;         // (B,H,hd,hd) f32

  char* ws = (char*)d_ws;
  size_t off = 0;
  auto alloc = [&](size_t bytes) -> char* {
    char* p = ws + off; off += (bytes + 255) & ~(size_t)255; return p;
  };
  const size_t DD2 = (size_t)D_EMB * D_EMB * sizeof(bf16);
  const size_t BSD2 = (size_t)BS * D_EMB * sizeof(bf16);
  const size_t BSD4 = (size_t)BS * D_EMB * sizeof(float);

  bf16* WqT = (bf16*)alloc(DD2);
  bf16* WkT = (bf16*)alloc(DD2);
  bf16* WvT = (bf16*)alloc(DD2);
  bf16* WoT = (bf16*)alloc(DD2);
  bf16* WgT = (bf16*)alloc(DD2);
  bf16* WpT = (bf16*)alloc(DD2);
  bf16* hT  = (bf16*)alloc((size_t)BH * HD * HD * sizeof(bf16));
  bf16* xpe = (bf16*)alloc(BSD2);
  bf16* xb  = (bf16*)alloc(BSD2);
  float* qf = (float*)alloc(BSD4);   // later reused as y = ret @ Wo
  float* kf = (float*)alloc(BSD4);   // later reused as gate GEMM out
  float* vf = (float*)alloc(BSD4);   // later reused as proj GEMM out
  bf16* qb  = (bf16*)alloc(BSD2);
  bf16* qdb = (bf16*)alloc(BSD2);
  bf16* kb  = (bf16*)alloc(BSD2);
  bf16* vT  = (bf16*)alloc(BSD2);
  bf16* kTw = (bf16*)alloc(BSD2);
  bf16* sc  = (bf16*)alloc((size_t)BH * SEQ * SEQ * sizeof(bf16));
  float* retf = (float*)alloc(BSD4);
  bf16* retb  = (bf16*)alloc(BSD2);
  float* x1f  = (float*)alloc(BSD4);
  bf16* x1b   = (bf16*)alloc(BSD2);
  int* seg    = (int*)alloc((size_t)BSZ * SEQ * sizeof(int));

  // 1. weight / hstate transposes + seg + PE
  int ebl = (D_EMB * D_EMB) / 256;
  k_transpose_bf16<<<dim3(ebl, 1), 256, 0, stream>>>(Wq, WqT, D_EMB, D_EMB);
  k_transpose_bf16<<<dim3(ebl, 1), 256, 0, stream>>>(Wk, WkT, D_EMB, D_EMB);
  k_transpose_bf16<<<dim3(ebl, 1), 256, 0, stream>>>(Wv, WvT, D_EMB, D_EMB);
  k_transpose_bf16<<<dim3(ebl, 1), 256, 0, stream>>>(Wo, WoT, D_EMB, D_EMB);
  k_transpose_bf16<<<dim3(ebl, 1), 256, 0, stream>>>(Wg, WgT, D_EMB, D_EMB);
  k_transpose_bf16<<<dim3(ebl, 1), 256, 0, stream>>>(Wp, WpT, D_EMB, D_EMB);
  k_transpose_bf16<<<dim3((HD * HD) / 256, BH), 256, 0, stream>>>(hstate, hT, HD, HD);
  k_seg<<<1, 32, 0, stream>>>(dones, seg);
  k_pe<<<BS, 256, 0, stream>>>(x, step, xpe, xb);

  // 2. q/k/v projections: (8192x1024)@(1024x1024), 64x16 = 1024 blocks of 128x64 tiles
  const int gemm_blocks = (BS / BM) * (D_EMB / BN);
  k_gemm_lds<<<gemm_blocks, 256, 0, stream>>>(xpe, D_EMB, WqT, D_EMB, qf, D_EMB, D_EMB, D_EMB);
  k_gemm_lds<<<gemm_blocks, 256, 0, stream>>>(xpe, D_EMB, WkT, D_EMB, kf, D_EMB, D_EMB, D_EMB);
  k_gemm_lds<<<gemm_blocks, 256, 0, stream>>>(xb,  D_EMB, WvT, D_EMB, vf, D_EMB, D_EMB, D_EMB);
  k_qkv_scatter<<<(BS * D_EMB) / 256, 256, 0, stream>>>(qf, kf, vf, seg,
                                                        qb, qdb, kb, vT, kTw);

  // 3. retention: scores -> inner -> cross -> next_h
  k_scores<<<dim3(128, BH), 256, 0, stream>>>(qb, kb, seg, sc);
  k_inner<<<dim3(32, BH), 256, 0, stream>>>(sc, vT, retf);
  k_cross<<<dim3(32, BH), 256, 0, stream>>>(qdb, hT, seg, retf);
  k_nexth<<<dim3(8, BH), 256, 0, stream>>>(kTw, vT, hstate, seg, nhout);

  // 4. output projection + first residual RMSNorm
  k_conv_bf16<<<(BS * D_EMB) / 256, 256, 0, stream>>>(retf, retb, (i64)BS * D_EMB);
  k_gemm_lds<<<gemm_blocks, 256, 0, stream>>>(retb, D_EMB, WoT, D_EMB, qf, D_EMB, D_EMB, D_EMB);
  k_rms1<<<BS, 256, 0, stream>>>(x, qf, wln1, x1f, x1b);

  // 5. SwiGLU FFN + final RMSNorm
  k_gemm_lds<<<gemm_blocks, 256, 0, stream>>>(x1b, D_EMB, WgT, D_EMB, kf, D_EMB, D_EMB, D_EMB);
  k_gemm_lds<<<gemm_blocks, 256, 0, stream>>>(x1b, D_EMB, WpT, D_EMB, vf, D_EMB, D_EMB, D_EMB);
  k_final<<<BS, 256, 0, stream>>>(x1f, kf, vf, wln2, out);
}
